// PointNet2_60473139528110
// MI455X (gfx1250) — matrix operand order repaired and checked by reading, not compiled, not run on hardware
//
#include <hip/hip_runtime.h>
#include <cstdint>
#include <cstddef>

typedef _Float16 half_t;
typedef __attribute__((ext_vector_type(16))) _Float16 v16h;
typedef __attribute__((ext_vector_type(8)))  float    v8f;

#define B_ 16
#define N_ 2048
#define NEG_INF (-3.402823466e38f)
#define BN_SCALE 0.9999950000374997f /* 1/sqrt(1+1e-5) */

// ---------------------------------------------------------------------------
// WMMA helpers (CDNA5 gfx1250, wave32, v_wmma_f32_16x16x32_f16)
// ---------------------------------------------------------------------------
__device__ __forceinline__ v8f wmma_f16(v16h a, v16h b, v8f c) {
  return __builtin_amdgcn_wmma_f32_16x16x32_f16(false, a, false, b, (short)0, c,
                                                false, false);
}

// A fragment (16x32, f16) gathered from row-major LDS buffer X (ld = ldx).
// Per ISA 7.12.2: lane l -> row M = l%16, half h = l/16;
// VGPR i<4: K = 2i + 8h ; VGPR i>=4: K = 16 + 2(i-4) + 8h.
__device__ __forceinline__ v16h load_a_frag(const half_t* X, int ldx,
                                            int rowBase, int kt, int lane) {
  int m = lane & 15, hh = lane >> 4;
  const half_t* row = X + (size_t)(rowBase + m) * ldx + kt * 32;
  v16h a;
#pragma unroll
  for (int i = 0; i < 8; ++i) {
    int c0 = (i < 4) ? (2 * i + 8 * hh) : (16 + 2 * (i - 4) + 8 * hh);
    a[2 * i]     = row[c0];
    a[2 * i + 1] = row[c0 + 1];
  }
  return a;
}

// B fragment (32x16, f16) from pre-swizzled weights: tile-major, 32 lanes x 16
// contiguous halves each -> two global_load_b128 per lane.
__device__ __forceinline__ v16h load_b_frag(const half_t* Wp, int ntiles,
                                            int kt, int nt, int lane) {
  const half_t* p = Wp + (((size_t)kt * ntiles + nt) * 32 + lane) * 16;
  return *(const v16h*)p;
}

// GEMM layer: Y(rows x ntiles*16) = relu(X(rows x ktiles*32) @ Wp + bias),
// result written to LDS as f16.  C/D layout: VGPR r -> row r + 8*(l/16),
// col = l%16.
__device__ __forceinline__ void gemm_to_lds(const half_t* X, int ldx,
                                            int rowTiles, int ktiles,
                                            const half_t* Wp, const half_t* Bp,
                                            int ntiles, half_t* Y, int ldy,
                                            int wave, int nwaves, int lane) {
  int m = lane & 15, hh = lane >> 4;
  int total = rowTiles * ntiles;
  for (int t = wave; t < total; t += nwaves) {
    int rt = t / ntiles, nt = t - rt * ntiles;
    v8f acc = {};
    for (int kt = 0; kt < ktiles; ++kt) {
      v16h a  = load_a_frag(X, ldx, rt * 16, kt, lane);
      v16h bf = load_b_frag(Wp, ntiles, kt, nt, lane);
      acc = wmma_f16(a, bf, acc);
    }
    float bv = (float)Bp[nt * 16 + m];
#pragma unroll
    for (int r = 0; r < 8; ++r) {
      float v = acc[r] + bv;
      v = v > 0.f ? v : 0.f;
      Y[(size_t)(rt * 16 + r + 8 * hh) * ldy + nt * 16 + m] = (half_t)v;
    }
  }
}

// ---------------------------------------------------------------------------
// Weight packing: fold BN, f32 -> f16, fragment-native tile layout.
// ---------------------------------------------------------------------------
__global__ void pack_layer_kernel(const float* __restrict__ W,
                                  const float* __restrict__ bias,
                                  const float* __restrict__ be,
                                  const float* __restrict__ g,
                                  int cin, int cout, int useBn,
                                  half_t* __restrict__ Wp,
                                  half_t* __restrict__ Bp) {
  int kt = blockIdx.x, nt = blockIdx.y, lane = threadIdx.x;
  int ntiles = gridDim.y;
  int m = lane & 15, hh = lane >> 4;
  int n = nt * 16 + m;
  float scale = 1.f, bb = 0.f;
  if (n < cout) {
    scale = useBn ? g[n] * BN_SCALE : 1.f;
    bb    = useBn ? bias[n] * scale + be[n] : bias[n];
  }
  half_t* dst = Wp + (((size_t)kt * ntiles + nt) * 32 + lane) * 16;
#pragma unroll
  for (int v = 0; v < 8; ++v) {
    int k0 = kt * 32 + ((v < 4) ? (2 * v + 8 * hh) : (16 + 2 * (v - 4) + 8 * hh));
    float w0 = (n < cout && k0 < cin)     ? W[(size_t)k0 * cout + n] * scale : 0.f;
    float w1 = (n < cout && k0 + 1 < cin) ? W[(size_t)(k0 + 1) * cout + n] * scale : 0.f;
    dst[2 * v]     = (half_t)w0;
    dst[2 * v + 1] = (half_t)w1;
  }
  if (kt == 0 && hh == 0) Bp[nt * 16 + m] = (half_t)(n < cout ? bb : 0.f);
}

// ---------------------------------------------------------------------------
// Input feature builder: feats0 = [xyz-mean, rgb, mean, state]  (41 ch)
// ---------------------------------------------------------------------------
__global__ void feats_kernel(const float* __restrict__ xyz,
                             const float* __restrict__ rgb,
                             const float* __restrict__ state,
                             const float* __restrict__ mask,
                             float* __restrict__ feats,
                             float* __restrict__ meanOut) {
  __shared__ float sx[256], sy[256], sz[256], sm[256];
  __shared__ float st[32];
  __shared__ float mean[3];
  int b = blockIdx.x, tid = threadIdx.x;
  float ax = 0, ay = 0, az = 0, am = 0;
  for (int i = tid; i < N_; i += 256) {
    float mm = mask[(size_t)b * N_ + i];
    const float* p = xyz + ((size_t)b * N_ + i) * 3;
    ax += p[0] * mm; ay += p[1] * mm; az += p[2] * mm; am += mm;
  }
  sx[tid] = ax; sy[tid] = ay; sz[tid] = az; sm[tid] = am;
  __syncthreads();
  for (int s = 128; s > 0; s >>= 1) {
    if (tid < s) { sx[tid] += sx[tid+s]; sy[tid] += sy[tid+s];
                   sz[tid] += sz[tid+s]; sm[tid] += sm[tid+s]; }
    __syncthreads();
  }
  if (tid == 0) {
    float d = fmaxf(sm[0], 1e-8f);
    mean[0] = sx[0] / d; mean[1] = sy[0] / d; mean[2] = sz[0] / d;
    meanOut[b * 3 + 0] = mean[0]; meanOut[b * 3 + 1] = mean[1];
    meanOut[b * 3 + 2] = mean[2];
  }
  if (tid < 32) st[tid] = state[b * 32 + tid];
  __syncthreads();
  for (int i = tid; i < N_; i += 256) {
    const float* p = xyz + ((size_t)b * N_ + i) * 3;
    const float* r = rgb + ((size_t)b * N_ + i) * 3;
    float* f = feats + ((size_t)b * N_ + i) * 41;
    f[0] = p[0] - mean[0]; f[1] = p[1] - mean[1]; f[2] = p[2] - mean[2];
    f[3] = r[0]; f[4] = r[1]; f[5] = r[2];
    f[6] = mean[0]; f[7] = mean[1]; f[8] = mean[2];
    for (int c = 0; c < 32; ++c) f[9 + c] = st[c];
  }
}

// ---------------------------------------------------------------------------
// Farthest point sampling (block per batch, deterministic start 0).
// ---------------------------------------------------------------------------
__global__ void fps_kernel(const float* __restrict__ xyz, int n, int npoint,
                           int* __restrict__ idx, float* __restrict__ new_xyz) {
  __shared__ float dist[N_];
  __shared__ float rv[256];
  __shared__ int   ri[256];
  int b = blockIdx.x, tid = threadIdx.x;
  for (int i = tid; i < n; i += 256) dist[i] = 1e10f;
  __syncthreads();
  int far = 0;
  for (int it = 0; it < npoint; ++it) {
    if (tid == 0) {
      idx[(size_t)b * npoint + it] = far;
      new_xyz[((size_t)b * npoint + it) * 3 + 0] = xyz[((size_t)b * n + far) * 3 + 0];
      new_xyz[((size_t)b * npoint + it) * 3 + 1] = xyz[((size_t)b * n + far) * 3 + 1];
      new_xyz[((size_t)b * npoint + it) * 3 + 2] = xyz[((size_t)b * n + far) * 3 + 2];
    }
    float cx = xyz[((size_t)b * n + far) * 3 + 0];
    float cy = xyz[((size_t)b * n + far) * 3 + 1];
    float cz = xyz[((size_t)b * n + far) * 3 + 2];
    float best = -1.f; int bi = 0;
    for (int i = tid; i < n; i += 256) {
      float dx = xyz[((size_t)b * n + i) * 3 + 0] - cx;
      float dy = xyz[((size_t)b * n + i) * 3 + 1] - cy;
      float dz = xyz[((size_t)b * n + i) * 3 + 2] - cz;
      float d = dx * dx + dy * dy + dz * dz;
      float m = fminf(dist[i], d);
      dist[i] = m;
      if (m > best) { best = m; bi = i; }
    }
    rv[tid] = best; ri[tid] = bi;
    __syncthreads();
    for (int s = 128; s > 0; s >>= 1) {
      if (tid < s) {
        if (rv[tid + s] > rv[tid] ||
            (rv[tid + s] == rv[tid] && ri[tid + s] < ri[tid])) {
          rv[tid] = rv[tid + s]; ri[tid] = ri[tid + s];
        }
      }
      __syncthreads();
    }
    far = ri[0];
    __syncthreads();
  }
}

// ---------------------------------------------------------------------------
// Ball query: first K in-range indices (ascending), pad with first hit.
// ---------------------------------------------------------------------------
__global__ void ball_query_kernel(const float* __restrict__ xyz, int n,
                                  const float* __restrict__ new_xyz, int S,
                                  float r2, int K, int* __restrict__ out,
                                  int total) {
  int t = blockIdx.x * blockDim.x + threadIdx.x;
  if (t >= total) return;
  int b = t / S, s = t - b * S;
  float cx = new_xyz[((size_t)b * S + s) * 3 + 0];
  float cy = new_xyz[((size_t)b * S + s) * 3 + 1];
  float cz = new_xyz[((size_t)b * S + s) * 3 + 2];
  int* o = out + ((size_t)b * S + s) * K;
  int cnt = 0, first = -1;
  for (int i = 0; i < n && cnt < K; ++i) {
    const float* p = xyz + ((size_t)b * n + i) * 3;
    float dx = p[0] - cx, dy = p[1] - cy, dz = p[2] - cz;
    float d = dx * dx + dy * dy + dz * dz;
    if (d <= r2) { if (first < 0) first = i; o[cnt++] = i; }
  }
  if (first < 0) first = 0;
  for (; cnt < K; ++cnt) o[cnt] = first;
}

// ---------------------------------------------------------------------------
// Fused SA branch: gather -> 3x (GEMM+BN+ReLU) -> max-pool over K samples.
// One block per sampled point, 4 waves. All GEMMs via WMMA.
// ---------------------------------------------------------------------------
template <int K, int CIN, int CINPAD, int C1, int C1P, int C2, int C2P, int C3,
          int C3P>
__global__ void __launch_bounds__(128)
sa_branch_kernel(const float* __restrict__ xyz, int n,
                 const float* __restrict__ feats,
                 const float* __restrict__ new_xyz, int S,
                 const int* __restrict__ gidx,
                 const half_t* __restrict__ W1, const half_t* __restrict__ Bb1,
                 const half_t* __restrict__ W2, const half_t* __restrict__ Bb2,
                 const half_t* __restrict__ W3, const half_t* __restrict__ Bb3,
                 float* __restrict__ out, int outStride, int outOff) {
  constexpr int C = CIN - 3;
  __shared__ half_t Xs[K * CINPAD];
  __shared__ half_t Ys[K * C1P];
  int b = blockIdx.y, s = blockIdx.x;
  int tid = threadIdx.x, lane = tid & 31, wave = tid >> 5;

  for (int i = tid; i < K * CINPAD; i += 128) Xs[i] = (half_t)0;
  for (int i = tid; i < K * C1P; i += 128) Ys[i] = (half_t)0;
  __syncthreads();

  // gather K neighbors: [feats | xyz - center]
  const int* gi = gidx + ((size_t)b * S + s) * K;
  float cx = new_xyz[((size_t)b * S + s) * 3 + 0];
  float cy = new_xyz[((size_t)b * S + s) * 3 + 1];
  float cz = new_xyz[((size_t)b * S + s) * 3 + 2];
  for (int j = wave; j < K; j += 4) {
    int id = gi[j];
    if (id >= n) id = n - 1;
    const float* f = feats + ((size_t)b * n + id) * C;
    for (int c = lane; c < C; c += 32) Xs[j * CINPAD + c] = (half_t)f[c];
    if (lane < 3) {
      float cc = (lane == 0) ? cx : (lane == 1) ? cy : cz;
      Xs[j * CINPAD + C + lane] =
          (half_t)(xyz[((size_t)b * n + id) * 3 + lane] - cc);
    }
  }
  __syncthreads();

  gemm_to_lds(Xs, CINPAD, K / 16, CINPAD / 32, W1, Bb1, C1P / 16, Ys, C1P,
              wave, 4, lane);
  __syncthreads();
  gemm_to_lds(Ys, C1P, K / 16, C1P / 32, W2, Bb2, C2P / 16, Xs, CINPAD, wave,
              4, lane);
  __syncthreads();

  // layer3 + max-pool over K directly on D fragments.
  int m = lane & 15;
  float* orow = out + ((size_t)b * S + s) * outStride + outOff;
  for (int nt = wave; nt < C3P / 16; nt += 4) {
    int col = nt * 16 + m;
    float bv = (float)Bb3[col];
    float cmax = NEG_INF;
    for (int rt = 0; rt < K / 16; ++rt) {
      v8f acc = {};
      for (int kt = 0; kt < C2P / 32; ++kt) {
        v16h a  = load_a_frag(Xs, CINPAD, rt * 16, kt, lane);
        v16h bf = load_b_frag(W3, C3P / 16, kt, nt, lane);
        acc = wmma_f16(a, bf, acc);
      }
#pragma unroll
      for (int r = 0; r < 8; ++r) {
        float v = acc[r] + bv;
        v = v > 0.f ? v : 0.f;
        if (v > cmax) cmax = v;
      }
    }
    float other = __shfl_xor(cmax, 16, 32);  // merge the two M halves
    cmax = fmaxf(cmax, other);
    if (lane < 16 && col < C3) orow[col] = cmax;
  }
}

// ---------------------------------------------------------------------------
// 3-NN (for feature propagation) + inverse-distance weights.
// ---------------------------------------------------------------------------
__global__ void knn3_kernel(const float* __restrict__ xyz1, int n1,
                            const float* __restrict__ xyz2, int n2,
                            int* __restrict__ idx3, float* __restrict__ w3,
                            int total) {
  int t = blockIdx.x * blockDim.x + threadIdx.x;
  if (t >= total) return;
  int b = t / n1, i = t - b * n1;
  float px = xyz1[((size_t)b * n1 + i) * 3 + 0];
  float py = xyz1[((size_t)b * n1 + i) * 3 + 1];
  float pz = xyz1[((size_t)b * n1 + i) * 3 + 2];
  float d0 = 3e38f, d1 = 3e38f, d2 = 3e38f;
  int j0 = 0, j1 = 0, j2 = 0;
  for (int j = 0; j < n2; ++j) {
    const float* q = xyz2 + ((size_t)b * n2 + j) * 3;
    float dx = px - q[0], dy = py - q[1], dz = pz - q[2];
    float d = dx * dx + dy * dy + dz * dz;
    if (d < d0)      { d2 = d1; j2 = j1; d1 = d0; j1 = j0; d0 = d; j0 = j; }
    else if (d < d1) { d2 = d1; j2 = j1; d1 = d; j1 = j; }
    else if (d < d2) { d2 = d; j2 = j; }
  }
  float w0 = 1.f / (d0 + 1e-8f), w1 = 1.f / (d1 + 1e-8f),
        w2 = 1.f / (d2 + 1e-8f);
  float sw = w0 + w1 + w2;
  size_t o = ((size_t)b * n1 + i) * 3;
  idx3[o] = j0; idx3[o + 1] = j1; idx3[o + 2] = j2;
  w3[o] = w0 / sw; w3[o + 1] = w1 / sw; w3[o + 2] = w2 / sw;
}

// xcat = concat(pts1, sum_k w_k * pts2[idx_k])
__global__ void xcat_kernel(const float* __restrict__ pts1, int ca,
                            const float* __restrict__ pts2, int cb, int n1,
                            int n2, const int* __restrict__ idx3,
                            const float* __restrict__ w3,
                            float* __restrict__ xcat, int total) {
  int t = blockIdx.x * blockDim.x + threadIdx.x;
  if (t >= total) return;
  int cw = ca + cb;
  int c = t % cw;
  int bi = t / cw;
  int i = bi % n1;
  int b = bi / n1;
  if (c < ca) {
    xcat[t] = pts1[((size_t)b * n1 + i) * ca + c];
  } else {
    int cc = c - ca;
    const int* id = idx3 + ((size_t)b * n1 + i) * 3;
    const float* w = w3 + ((size_t)b * n1 + i) * 3;
    float s = 0.f;
#pragma unroll
    for (int k = 0; k < 3; ++k)
      s += w[k] * pts2[((size_t)b * n2 + id[k]) * cb + cc];
    xcat[t] = s;
  }
}

// ---------------------------------------------------------------------------
// Generic per-point 2-layer MLP (FP stages + head), WMMA.
// ---------------------------------------------------------------------------
template <int ROWS, int CIN, int CINPAD, int C1, int C1P, int C2, int C2P,
          bool RELU2>
__global__ void __launch_bounds__(128)
mlp_points_kernel(const float* __restrict__ Xin, int n1,
                  const half_t* __restrict__ W1, const half_t* __restrict__ Bb1,
                  const half_t* __restrict__ W2, const half_t* __restrict__ Bb2,
                  float* __restrict__ out) {
  __shared__ half_t Xs[ROWS * CINPAD];
  __shared__ half_t Ys[ROWS * C1P];
  int b = blockIdx.y;
  int r0 = blockIdx.x * ROWS;
  int tid = threadIdx.x, lane = tid & 31, wave = tid >> 5;

  for (int i = tid; i < ROWS * CINPAD; i += 128) Xs[i] = (half_t)0;
  for (int i = tid; i < ROWS * C1P; i += 128) Ys[i] = (half_t)0;
  __syncthreads();
  for (int j = wave; j < ROWS; j += 4) {
    int row = r0 + j;
    if (row < n1) {
      const float* src = Xin + ((size_t)b * n1 + row) * CIN;
      for (int c = lane; c < CIN; c += 32) Xs[j * CINPAD + c] = (half_t)src[c];
    }
  }
  __syncthreads();

  gemm_to_lds(Xs, CINPAD, ROWS / 16, CINPAD / 32, W1, Bb1, C1P / 16, Ys, C1P,
              wave, 4, lane);
  __syncthreads();

  int m = lane & 15, hh = lane >> 4;
  int rowTiles = ROWS / 16, ntiles = C2P / 16, total = rowTiles * ntiles;
  for (int t = wave; t < total; t += 4) {
    int rt = t / ntiles, nt = t - rt * ntiles;
    v8f acc = {};
    for (int kt = 0; kt < C1P / 32; ++kt) {
      v16h a  = load_a_frag(Ys, C1P, rt * 16, kt, lane);
      v16h bf = load_b_frag(W2, ntiles, kt, nt, lane);
      acc = wmma_f16(a, bf, acc);
    }
    int col = nt * 16 + m;
    float bv = (float)Bb2[col];
#pragma unroll
    for (int r = 0; r < 8; ++r) {
      int row = r0 + rt * 16 + r + 8 * hh;
      float v = acc[r] + bv;
      if (RELU2) v = v > 0.f ? v : 0.f;
      if (row < n1 && col < C2) out[((size_t)b * n1 + row) * C2 + col] = v;
    }
  }
}

// ---------------------------------------------------------------------------
// Global max/mean aggregation over all N points.
// ---------------------------------------------------------------------------
__global__ void aggregate_kernel(const float* __restrict__ l0,
                                 const float* __restrict__ mask,
                                 float* __restrict__ gf) {
  int t = blockIdx.x * blockDim.x + threadIdx.x;
  if (t >= B_ * 256) return;
  int b = t / 256, c = t - b * 256;
  if (c < 128) {
    float mx = NEG_INF;
    for (int i = 0; i < N_; ++i)
      if (mask[(size_t)b * N_ + i] > 0.5f) {
        float v = l0[((size_t)b * N_ + i) * 256 + c];
        mx = fmaxf(mx, v);
      }
    gf[(size_t)b * 256 + c] = mx;
  } else {
    float s = 0.f, sm = 0.f;
    for (int i = 0; i < N_; ++i) {
      float mm = mask[(size_t)b * N_ + i];
      s += l0[((size_t)b * N_ + i) * 256 + c] * mm;
      sm += mm;
    }
    gf[(size_t)b * 256 + c] = s / fmaxf(sm, 1e-8f);
  }
}

// ---------------------------------------------------------------------------
// Host driver
// ---------------------------------------------------------------------------
extern "C" void kernel_launch(void* const* d_in, const int* in_sizes, int n_in,
                              void* d_out, int out_size, void* d_ws,
                              size_t ws_size, hipStream_t stream) {
  (void)in_sizes; (void)n_in; (void)out_size; (void)ws_size;
  const float* xyz0 = (const float*)d_in[0];
  const float* rgb  = (const float*)d_in[1];
  const float* stt  = (const float*)d_in[2];
  const float* mask = (const float*)d_in[3];
  auto leaf = [&](int i) { return (const float*)d_in[i]; };

  char* wsb = (char*)d_ws;
  size_t off = 0;
  auto alloc = [&](size_t bytes) -> void* {
    off = (off + 255) & ~(size_t)255;
    void* p = wsb + off;
    off += bytes;
    return p;
  };
  auto pad32 = [](int x) { return (x + 31) & ~31; };

  // ---------------- scratch ----------------
  float* meanb  = (float*)alloc((size_t)B_ * 3 * sizeof(float));
  float* feats0 = (float*)alloc((size_t)B_ * N_ * 41 * sizeof(float));
  int* i1 = (int*)alloc((size_t)B_ * 512 * sizeof(int));
  int* i2 = (int*)alloc((size_t)B_ * 128 * sizeof(int));
  int* i3 = (int*)alloc((size_t)B_ * 64 * sizeof(int));
  int* i4 = (int*)alloc((size_t)B_ * 16 * sizeof(int));
  float* x1 = (float*)alloc((size_t)B_ * 512 * 3 * sizeof(float));
  float* x2 = (float*)alloc((size_t)B_ * 128 * 3 * sizeof(float));
  float* x3 = (float*)alloc((size_t)B_ * 64 * 3 * sizeof(float));
  float* x4 = (float*)alloc((size_t)B_ * 16 * 3 * sizeof(float));
  int* g00 = (int*)alloc((size_t)B_ * 512 * 16 * sizeof(int));
  int* g01 = (int*)alloc((size_t)B_ * 512 * 32 * sizeof(int));
  int* g10 = (int*)alloc((size_t)B_ * 128 * 16 * sizeof(int));
  int* g11 = (int*)alloc((size_t)B_ * 128 * 32 * sizeof(int));
  int* g20 = (int*)alloc((size_t)B_ * 64 * 16 * sizeof(int));
  int* g21 = (int*)alloc((size_t)B_ * 64 * 32 * sizeof(int));
  int* g30 = (int*)alloc((size_t)B_ * 16 * 16 * sizeof(int));
  int* g31 = (int*)alloc((size_t)B_ * 16 * 32 * sizeof(int));
  float* f1 = (float*)alloc((size_t)B_ * 512 * 96 * sizeof(float));
  float* f2 = (float*)alloc((size_t)B_ * 128 * 256 * sizeof(float));
  float* f3 = (float*)alloc((size_t)B_ * 64 * 512 * sizeof(float));
  float* f4 = (float*)alloc((size_t)B_ * 16 * 1024 * sizeof(float));
  int*   knnI = (int*)alloc((size_t)B_ * N_ * 3 * sizeof(int));
  float* knnW = (float*)alloc((size_t)B_ * N_ * 3 * sizeof(float));
  float* xcat = (float*)alloc((size_t)B_ * N_ * 256 * sizeof(float)); // max
  float* o0 = (float*)alloc((size_t)B_ * 64 * 256 * sizeof(float));
  float* o1 = (float*)alloc((size_t)B_ * 128 * 256 * sizeof(float));
  float* o2 = (float*)alloc((size_t)B_ * 512 * 256 * sizeof(float));
  float* o3 = (float*)alloc((size_t)B_ * N_ * 256 * sizeof(float));
  float* gf = (float*)alloc((size_t)B_ * 256 * sizeof(float));
  (void)meanb; (void)i1; (void)i2; (void)i3; (void)i4;

  // ---------------- weight packing (BN folded, f16 fragments) --------------
  auto packLayer = [&](int wi, int bIdx, int beIdx, int gIdx, int cin,
                       int cout, int useBn, half_t*& Wp, half_t*& Bp) {
    int kT = pad32(cin) / 32, nT = pad32(cout) / 16;
    Wp = (half_t*)alloc((size_t)pad32(cin) * pad32(cout) * sizeof(half_t));
    Bp = (half_t*)alloc((size_t)pad32(cout) * sizeof(half_t));
    pack_layer_kernel<<<dim3(kT, nT), 32, 0, stream>>>(
        leaf(wi), leaf(bIdx),
        beIdx >= 0 ? leaf(beIdx) : (const float*)nullptr,
        gIdx >= 0 ? leaf(gIdx) : (const float*)nullptr, cin, cout, useBn, Wp,
        Bp);
  };

  // pytree flattening: dicts sorted by key -> leaves:
  // fp0..fp3 (2 layers x {W,b,be,g}), head {W1,W2,b1,b2}, sa0..sa3
  // (2 branches x 3 layers x {W,b,be,g}); inputs occupy d_in[0..3].
  const int sa_cin[4] = {44, 99, 259, 515};
  const int sa_ch[4][2][3] = {
      {{16, 16, 32}, {32, 32, 64}},
      {{64, 64, 128}, {64, 96, 128}},
      {{128, 196, 256}, {128, 196, 256}},
      {{256, 256, 512}, {256, 384, 512}}};
  half_t *saW[4][2][3], *saB[4][2][3];
  for (int si = 0; si < 4; ++si)
    for (int bi = 0; bi < 2; ++bi) {
      int cin = sa_cin[si];
      for (int l = 0; l < 3; ++l) {
        int co = sa_ch[si][bi][l];
        int wI = 40 + si * 24 + bi * 12 + l * 4;
        packLayer(wI, wI + 1, wI + 2, wI + 3, cin, co, 1, saW[si][bi][l],
                  saB[si][bi][l]);
        cin = co;
      }
    }
  const int fp_cin[4] = {1536, 512, 352, 256};
  half_t *fpW[4][2], *fpB[4][2];
  for (int f = 0; f < 4; ++f) {
    int w0 = 4 + f * 8;
    packLayer(w0, w0 + 1, w0 + 2, w0 + 3, fp_cin[f], 256, 1, fpW[f][0],
              fpB[f][0]);
    packLayer(w0 + 4, w0 + 5, w0 + 6, w0 + 7, 256, 256, 1, fpW[f][1],
              fpB[f][1]);
  }
  half_t *hW[2], *hB[2];
  packLayer(36, 38, -1, -1, 256, 512, 0, hW[0], hB[0]);
  packLayer(37, 39, -1, -1, 512, 256, 0, hW[1], hB[1]);

  auto ball = [&](const float* xyz, int n, const float* nx, int S, float r,
                  int K, int* out) {
    int tot = B_ * S;
    ball_query_kernel<<<(tot + 255) / 256, 256, 0, stream>>>(xyz, n, nx, S,
                                                             r * r, K, out,
                                                             tot);
  };

  // ---------------- input features ----------------
  feats_kernel<<<B_, 256, 0, stream>>>(xyz0, rgb, stt, mask, feats0, meanb);

  // ---------------- SA stage 1 (2048 -> 512) ----------------
  fps_kernel<<<B_, 256, 0, stream>>>(xyz0, N_, 512, i1, x1);
  ball(xyz0, N_, x1, 512, 0.05f, 16, g00);
  ball(xyz0, N_, x1, 512, 0.10f, 32, g01);
  sa_branch_kernel<16, 44, 64, 16, 32, 16, 32, 32, 32>
      <<<dim3(512, B_), 128, 0, stream>>>(xyz0, N_, feats0, x1, 512, g00,
          saW[0][0][0], saB[0][0][0], saW[0][0][1], saB[0][0][1],
          saW[0][0][2], saB[0][0][2], f1, 96, 0);
  sa_branch_kernel<32, 44, 64, 32, 32, 32, 32, 64, 64>
      <<<dim3(512, B_), 128, 0, stream>>>(xyz0, N_, feats0, x1, 512, g01,
          saW[0][1][0], saB[0][1][0], saW[0][1][1], saB[0][1][1],
          saW[0][1][2], saB[0][1][2], f1, 96, 32);

  // ---------------- SA stage 2 (512 -> 128) ----------------
  fps_kernel<<<B_, 256, 0, stream>>>(x1, 512, 128, i2, x2);
  ball(x1, 512, x2, 128, 0.10f, 16, g10);
  ball(x1, 512, x2, 128, 0.20f, 32, g11);
  sa_branch_kernel<16, 99, 128, 64, 64, 64, 64, 128, 128>
      <<<dim3(128, B_), 128, 0, stream>>>(x1, 512, f1, x2, 128, g10,
          saW[1][0][0], saB[1][0][0], saW[1][0][1], saB[1][0][1],
          saW[1][0][2], saB[1][0][2], f2, 256, 0);
  sa_branch_kernel<32, 99, 128, 64, 64, 96, 96, 128, 128>
      <<<dim3(128, B_), 128, 0, stream>>>(x1, 512, f1, x2, 128, g11,
          saW[1][1][0], saB[1][1][0], saW[1][1][1], saB[1][1][1],
          saW[1][1][2], saB[1][1][2], f2, 256, 128);

  // ---------------- SA stage 3 (128 -> 64) ----------------
  fps_kernel<<<B_, 256, 0, stream>>>(x2, 128, 64, i3, x3);
  ball(x2, 128, x3, 64, 0.20f, 16, g20);
  ball(x2, 128, x3, 64, 0.40f, 32, g21);
  sa_branch_kernel<16, 259, 288, 128, 128, 196, 224, 256, 256>
      <<<dim3(64, B_), 128, 0, stream>>>(x2, 128, f2, x3, 64, g20,
          saW[2][0][0], saB[2][0][0], saW[2][0][1], saB[2][0][1],
          saW[2][0][2], saB[2][0][2], f3, 512, 0);
  sa_branch_kernel<32, 259, 288, 128, 128, 196, 224, 256, 256>
      <<<dim3(64, B_), 128, 0, stream>>>(x2, 128, f2, x3, 64, g21,
          saW[2][1][0], saB[2][1][0], saW[2][1][1], saB[2][1][1],
          saW[2][1][2], saB[2][1][2], f3, 512, 256);

  // ---------------- SA stage 4 (64 -> 16) ----------------
  fps_kernel<<<B_, 256, 0, stream>>>(x3, 64, 16, i4, x4);
  ball(x3, 64, x4, 16, 0.40f, 16, g30);
  ball(x3, 64, x4, 16, 0.80f, 32, g31);
  sa_branch_kernel<16, 515, 544, 256, 256, 256, 256, 512, 512>
      <<<dim3(16, B_), 128, 0, stream>>>(x3, 64, f3, x4, 16, g30,
          saW[3][0][0], saB[3][0][0], saW[3][0][1], saB[3][0][1],
          saW[3][0][2], saB[3][0][2], f4, 1024, 0);
  sa_branch_kernel<32, 515, 544, 256, 256, 384, 384, 512, 512>
      <<<dim3(16, B_), 128, 0, stream>>>(x3, 64, f3, x4, 16, g31,
          saW[3][1][0], saB[3][1][0], saW[3][1][1], saB[3][1][1],
          saW[3][1][2], saB[3][1][2], f4, 1024, 512);

  // ---------------- FP0: (64 <- 16), cin 512+1024 ----------------
  knn3_kernel<<<(B_ * 64 + 255) / 256, 256, 0, stream>>>(x3, 64, x4, 16, knnI,
                                                         knnW, B_ * 64);
  {
    int tot = B_ * 64 * 1536;
    xcat_kernel<<<(tot + 255) / 256, 256, 0, stream>>>(f3, 512, f4, 1024, 64,
                                                       16, knnI, knnW, xcat,
                                                       tot);
  }
  mlp_points_kernel<16, 1536, 1536, 256, 256, 256, 256, true>
      <<<dim3(4, B_), 128, 0, stream>>>(xcat, 64, fpW[0][0], fpB[0][0],
                                        fpW[0][1], fpB[0][1], o0);

  // ---------------- FP1: (128 <- 64), cin 256+256 ----------------
  knn3_kernel<<<(B_ * 128 + 255) / 256, 256, 0, stream>>>(x2, 128, x3, 64,
                                                          knnI, knnW, B_ * 128);
  {
    int tot = B_ * 128 * 512;
    xcat_kernel<<<(tot + 255) / 256, 256, 0, stream>>>(f2, 256, o0, 256, 128,
                                                       64, knnI, knnW, xcat,
                                                       tot);
  }
  mlp_points_kernel<16, 512, 512, 256, 256, 256, 256, true>
      <<<dim3(8, B_), 128, 0, stream>>>(xcat, 128, fpW[1][0], fpB[1][0],
                                        fpW[1][1], fpB[1][1], o1);

  // ---------------- FP2: (512 <- 128), cin 96+256 ----------------
  knn3_kernel<<<(B_ * 512 + 255) / 256, 256, 0, stream>>>(x1, 512, x2, 128,
                                                          knnI, knnW, B_ * 512);
  {
    int tot = B_ * 512 * 352;
    xcat_kernel<<<(tot + 255) / 256, 256, 0, stream>>>(f1, 96, o1, 256, 512,
                                                       128, knnI, knnW, xcat,
                                                       tot);
  }
  mlp_points_kernel<16, 352, 352, 256, 256, 256, 256, true>
      <<<dim3(32, B_), 128, 0, stream>>>(xcat, 512, fpW[2][0], fpB[2][0],
                                         fpW[2][1], fpB[2][1], o2);

  // ---------------- FP3: (2048 <- 512), cin 256 ----------------
  knn3_kernel<<<(B_ * N_ + 255) / 256, 256, 0, stream>>>(xyz0, N_, x1, 512,
                                                         knnI, knnW, B_ * N_);
  {
    int tot = B_ * N_ * 256;
    xcat_kernel<<<(tot + 255) / 256, 256, 0, stream>>>(
        (const float*)nullptr, 0, o2, 256, N_, 512, knnI, knnW, xcat, tot);
  }
  mlp_points_kernel<16, 256, 256, 256, 256, 256, 256, true>
      <<<dim3(128, B_), 128, 0, stream>>>(xcat, N_, fpW[3][0], fpB[3][0],
                                          fpW[3][1], fpB[3][1], o3);

  // ---------------- aggregate + head ----------------
  aggregate_kernel<<<(B_ * 256 + 255) / 256, 256, 0, stream>>>(o3, mask, gf);
  mlp_points_kernel<16, 256, 256, 512, 512, 256, 256, false>
      <<<dim3(1, 1), 128, 0, stream>>>(gf, 16, hW[0], hB[0], hW[1], hB[1],
                                       (float*)d_out);
}
